// Transformer_22076131901661
// MI455X (gfx1250) — matrix operand order repaired and checked by reading, not compiled, not run on hardware
//
#include <hip/hip_runtime.h>
#include <cstdint>
#include <cstddef>

#define SEQ    4096
#define NIN    1024
#define NHD    128
#define NOUTD  1024
#define NHEADS 8
#define DPH    1280      // per-head: 2*NH + NOUT
#define DTOT   10240

typedef __attribute__((ext_vector_type(16))) __bf16 v16bf;
typedef __attribute__((ext_vector_type(8)))  float  v8f;
typedef unsigned int v4u __attribute__((ext_vector_type(4)));
typedef int          v8i __attribute__((ext_vector_type(8)));
typedef int          v4i __attribute__((ext_vector_type(4)));

union FragAB { uint4 q[2]; v16bf v; };
union Frag8  { v8f v; float f[8]; };
union Pack8  { unsigned short s[8]; uint4 u; };

// LDS tile geometry: 128 rows x 32 bf16 (64B) padded by TDM to 80B stride (bank-conflict free)
#define TROWS     128
#define TK        32
#define ROW_US    40            // 80 bytes / 2 = elements per padded row
#define TILE_US   (TROWS * ROW_US)

__device__ __forceinline__ unsigned short f2bf(float x) {
  union { float f; unsigned u; } c; c.f = x;
  unsigned r = c.u + 0x7FFFu + ((c.u >> 16) & 1u);   // round-to-nearest-even
  return (unsigned short)(r >> 16);
}

__device__ __forceinline__ unsigned lds_off(const void* p) {
  return (unsigned)(unsigned long long)(__attribute__((address_space(3))) const void*)p;
}

// Issue a TDM 2-D tile load: TK x TROWS bf16 tile, row stride `stride_elems`,
// LDS destination padded 16B per 64B row (pad_interval=3 -> 16 DWORDs, pad_amount=3 -> 4 DWORDs).
__device__ __forceinline__ void tdm_load_tile(unsigned lds_byte, const unsigned short* g,
                                              unsigned stride_elems) {
  unsigned long long ga = (unsigned long long)(size_t)g;
  v4u g0;
  g0[0] = 1u;                                              // count=1, user descriptor
  g0[1] = lds_byte;                                        // lds_addr
  g0[2] = (unsigned)(ga & 0xFFFFFFFFu);                    // global_addr[31:0]
  g0[3] = (unsigned)((ga >> 32) & 0x01FFFFFFu) | (2u << 30); // global_addr[56:32] | type=2
  v8i g1;
  const unsigned td0 = 1u << 20, td1 = 1u << 20;           // generous tensor dims (tile in-bounds)
  g1[0] = (int)((1u << 16) | (1u << 20) | (3u << 22) | (3u << 25)); // data_size=2B, pad cfg
  g1[1] = (int)((td0 & 0xFFFFu) << 16);                    // abar=0 | tensor_dim0 lo
  g1[2] = (int)(((td0 >> 16) & 0xFFFFu) | ((td1 & 0xFFFFu) << 16));
  g1[3] = (int)(((td1 >> 16) & 0xFFFFu) | ((unsigned)TK << 16)); // tensor_dim1 hi | tile_dim0
  g1[4] = (int)TROWS;                                      // tile_dim1 | tile_dim2=0
  g1[5] = (int)stride_elems;                               // tensor_dim0_stride[31:0]
  g1[6] = 0;                                               // stride0 hi | stride1 lo
  g1[7] = 0;                                               // stride1 hi
  v4i z4 = (v4i)(0);
  v8i z8 = (v8i)(0);
  __builtin_amdgcn_tensor_load_to_lds(g0, g1, z4, z4, z8, 0);
}

// ---- fragment loads from global (K-contiguous source rows) ----
__device__ __forceinline__ v16bf ldA(const unsigned short* p, int row0, int ld, int k0, int lane) {
  int m = lane & 15, kh = lane >> 4;
  const unsigned short* r = p + (size_t)(row0 + m) * ld + (k0 + 8 * kh);
  FragAB f;
  f.q[0] = *reinterpret_cast<const uint4*>(r);
  f.q[1] = *reinterpret_cast<const uint4*>(r + 16);
  return f.v;
}
__device__ __forceinline__ v16bf ldB(const unsigned short* p, int col0, int ld, int k0, int lane) {
  int n = lane & 15, kh = lane >> 4;
  const unsigned short* r = p + (size_t)(col0 + n) * ld + (k0 + 8 * kh);
  FragAB f;
  f.q[0] = *reinterpret_cast<const uint4*>(r);
  f.q[1] = *reinterpret_cast<const uint4*>(r + 16);
  return f.v;
}

// ---- fragment loads from padded LDS tiles ----
__device__ __forceinline__ v16bf ldA_lds(const unsigned short* t, int row, int lane) {
  int m = lane & 15, kh = lane >> 4;
  const unsigned short* r = t + (row + m) * ROW_US + 8 * kh;
  FragAB f;
  f.q[0] = *reinterpret_cast<const uint4*>(r);
  f.q[1] = *reinterpret_cast<const uint4*>(r + 16);
  return f.v;
}
__device__ __forceinline__ v16bf ldB_lds(const unsigned short* t, int row, int lane) {
  int n = lane & 15, kh = lane >> 4;
  const unsigned short* r = t + (row + n) * ROW_US + 8 * kh;
  FragAB f;
  f.q[0] = *reinterpret_cast<const uint4*>(r);
  f.q[1] = *reinterpret_cast<const uint4*>(r + 16);
  return f.v;
}

__device__ __forceinline__ v8f wmma_bf16(v16bf a, v16bf b, v8f c) {
  return __builtin_amdgcn_wmma_f32_16x16x32_bf16(false, a, false, b, (short)0, c, false, false);
}

// ---------------- kernel 0: f32 -> bf16 convert (vectorized x4) ----------------
__global__ __launch_bounds__(256) void cvt_f32_bf16(const float* __restrict__ src,
                                                    unsigned short* __restrict__ dst, int n4) {
  int i = blockIdx.x * 256 + threadIdx.x;
  if (i >= n4) return;
  float4 v = reinterpret_cast<const float4*>(src)[i];
  union { unsigned short s[4]; uint2 u; } o;
  o.s[0] = f2bf(v.x); o.s[1] = f2bf(v.y); o.s[2] = f2bf(v.z); o.s[3] = f2bf(v.w);
  reinterpret_cast<uint2*>(dst)[i] = o.u;
}

// ---------------- kernel 1: fused QKV GEMM  Y = x @ W^T + b ----------------
// Block tile 128x128, 8 waves (2 M x 4 N). TDM double-buffered LDS staging of both operands.
__global__ __launch_bounds__(256) void qkv_gemm(const unsigned short* __restrict__ xb,
                                                const unsigned short* __restrict__ Wb,
                                                const float* __restrict__ bias,
                                                unsigned short* __restrict__ Kb,
                                                unsigned short* __restrict__ Qb,
                                                unsigned short* __restrict__ Vt) {
  __shared__ __align__(16) unsigned short As[2][TILE_US];
  __shared__ __align__(16) unsigned short Bs[2][TILE_US];
  const int lane = threadIdx.x & 31, wid = threadIdx.x >> 5;
  const int wm = wid >> 2, wn = wid & 3;
  const int m0 = blockIdx.x * 128;
  const int n0 = blockIdx.y * 128;

  Frag8 acc[4][2];
#pragma unroll
  for (int mi = 0; mi < 4; ++mi)
#pragma unroll
    for (int ni = 0; ni < 2; ++ni) acc[mi][ni].v = (v8f)(0.f);

  if (wid == 0) {
    tdm_load_tile(lds_off(&As[0][0]), xb + (size_t)m0 * NIN, NIN);
    tdm_load_tile(lds_off(&Bs[0][0]), Wb + (size_t)n0 * NIN, NIN);
  }

  const int steps = NIN / TK;   // 32
  for (int s = 0; s < steps; ++s) {
    if (wid == 0) __builtin_amdgcn_s_wait_tensorcnt(0);
    __syncthreads();            // buffer `cur` ready; previous reads of `cur^1` done
    const int cur = s & 1;
    if (wid == 0 && s + 1 < steps) {
      const int k0 = (s + 1) * TK;
      tdm_load_tile(lds_off(&As[cur ^ 1][0]), xb + (size_t)m0 * NIN + k0, NIN);
      tdm_load_tile(lds_off(&Bs[cur ^ 1][0]), Wb + (size_t)n0 * NIN + k0, NIN);
    }
    v16bf a[4], bm[2];
#pragma unroll
    for (int mi = 0; mi < 4; ++mi) a[mi] = ldA_lds(&As[cur][0], wm * 64 + 16 * mi, lane);
#pragma unroll
    for (int ni = 0; ni < 2; ++ni) bm[ni] = ldB_lds(&Bs[cur][0], wn * 32 + 16 * ni, lane);
#pragma unroll
    for (int mi = 0; mi < 4; ++mi)
#pragma unroll
      for (int ni = 0; ni < 2; ++ni) acc[mi][ni].v = wmma_bf16(a[mi], bm[ni], acc[mi][ni].v);
  }

  // C layout: lane holds column n=lane&15; VGPR r holds row m = r + 8*(lane>>4).
  const int n = lane & 15, mh = lane >> 4;
#pragma unroll
  for (int ni = 0; ni < 2; ++ni) {
    const int d = n0 + wn * 32 + ni * 16 + n;  // output feature
    const int h = d / DPH, c = d % DPH;        // head, channel-in-head
    const float bv = bias[d];
#pragma unroll
    for (int mi = 0; mi < 4; ++mi) {
      const int s0 = m0 + wm * 64 + 16 * mi + 8 * mh;  // 8 consecutive seq rows
      if (c < NHD) {
        const float scale = 0.08838834764831845f;      // 1/sqrt(128)
#pragma unroll
        for (int r = 0; r < 8; ++r)
          Kb[(size_t)(s0 + r) * (NHEADS * NHD) + h * NHD + c] =
              f2bf((acc[mi][ni].f[r] + bv) * scale);
      } else if (c < 2 * NHD) {
#pragma unroll
        for (int r = 0; r < 8; ++r)
          Qb[(size_t)(s0 + r) * (NHEADS * NHD) + h * NHD + (c - NHD)] =
              f2bf(acc[mi][ni].f[r] + bv);
      } else {
        Pack8 pk;
#pragma unroll
        for (int r = 0; r < 8; ++r) pk.s[r] = f2bf(acc[mi][ni].f[r] + bv);
        *reinterpret_cast<uint4*>(&Vt[((size_t)h * NOUTD + (c - 2 * NHD)) * SEQ + s0]) = pk.u;
      }
    }
  }
}

// ---------------- kernel 2: per-(i-tile, head) softmax, writes P[h][i][j] bf16 ----------------
__global__ __launch_bounds__(256) void attn_softmax(const unsigned short* __restrict__ Kb,
                                                    const unsigned short* __restrict__ Qb,
                                                    unsigned short* __restrict__ P) {
  const int lane = threadIdx.x & 31, w = threadIdx.x >> 5;
  const int i0 = blockIdx.x * 16;
  const int h = blockIdx.y;
  const int kcol = h * NHD;
  const int ldq = NHEADS * NHD;

  v16bf Bk[4];  // K-tile as B operand, persistent
#pragma unroll
  for (int kc = 0; kc < 4; ++kc) Bk[kc] = ldB(Kb, i0, ldq, kcol + 32 * kc, lane);

  const float NEG = -3.0e38f;
  float m_loc = NEG, l_loc = 0.f;

  // phase 1: online stats over this wave's j stripes
  for (int jb = w * 32; jb < SEQ; jb += 256) {
    if (jb + 256 < SEQ)
      __builtin_prefetch(&Qb[(size_t)(jb + 256 + (lane & 15)) * ldq + kcol], 0, 1);
    Frag8 s0, s1; s0.v = (v8f)(0.f); s1.v = (v8f)(0.f);
#pragma unroll
    for (int kc = 0; kc < 4; ++kc) {
      v16bf a0 = ldA(Qb, jb,      ldq, kcol + 32 * kc, lane);
      v16bf a1 = ldA(Qb, jb + 16, ldq, kcol + 32 * kc, lane);
      s0.v = wmma_bf16(a0, Bk[kc], s0.v);
      s1.v = wmma_bf16(a1, Bk[kc], s1.v);
    }
    float bmax = NEG;
#pragma unroll
    for (int r = 0; r < 8; ++r) bmax = fmaxf(bmax, fmaxf(s0.f[r], s1.f[r]));
    bmax = fmaxf(bmax, __shfl_xor(bmax, 16));
    const float mnew = fmaxf(m_loc, bmax);
    float rs = 0.f;
#pragma unroll
    for (int r = 0; r < 8; ++r) rs += __expf(s0.f[r] - mnew) + __expf(s1.f[r] - mnew);
    rs += __shfl_xor(rs, 16);
    l_loc = l_loc * __expf(m_loc - mnew) + rs;
    m_loc = mnew;
  }

  // merge stats across the 8 waves (per column i = lane&15)
  __shared__ float sm[8][16], sl[8][16];
  if (lane < 16) { sm[w][lane] = m_loc; sl[w][lane] = l_loc; }
  __syncthreads();
  const int i = lane & 15;
  float mg = NEG;
#pragma unroll
  for (int ww = 0; ww < 8; ++ww) mg = fmaxf(mg, sm[ww][i]);
  float lg = 0.f;
#pragma unroll
  for (int ww = 0; ww < 8; ++ww) lg += sl[ww][i] * __expf(sm[ww][i] - mg);
  const float inv = 1.0f / lg;

  // phase 2: recompute logits, write normalized P
  const int kh = lane >> 4;
  for (int jb = w * 32; jb < SEQ; jb += 256) {
    Frag8 s0, s1; s0.v = (v8f)(0.f); s1.v = (v8f)(0.f);
#pragma unroll
    for (int kc = 0; kc < 4; ++kc) {
      v16bf a0 = ldA(Qb, jb,      ldq, kcol + 32 * kc, lane);
      v16bf a1 = ldA(Qb, jb + 16, ldq, kcol + 32 * kc, lane);
      s0.v = wmma_bf16(a0, Bk[kc], s0.v);
      s1.v = wmma_bf16(a1, Bk[kc], s1.v);
    }
    Pack8 p0, p1;
#pragma unroll
    for (int r = 0; r < 8; ++r) {
      p0.s[r] = f2bf(__expf(s0.f[r] - mg) * inv);
      p1.s[r] = f2bf(__expf(s1.f[r] - mg) * inv);
    }
    const size_t base = ((size_t)h * SEQ + (i0 + i)) * SEQ;
    *reinterpret_cast<uint4*>(&P[base + jb + 8 * kh])      = p0.u;
    *reinterpret_cast<uint4*>(&P[base + jb + 16 + 8 * kh]) = p1.u;
  }
}

// ---------------- kernel 3: O = sum_h P_h @ V_h (TDM-staged streaming GEMM) ----------------
__global__ __launch_bounds__(256) void av_gemm(const unsigned short* __restrict__ P,
                                               const unsigned short* __restrict__ Vt,
                                               float* __restrict__ O) {
  __shared__ __align__(16) unsigned short As[2][TILE_US];
  __shared__ __align__(16) unsigned short Bs[2][TILE_US];
  const int lane = threadIdx.x & 31, wid = threadIdx.x >> 5;
  const int wm = wid >> 2, wn = wid & 3;
  const int m0 = blockIdx.x * 128;
  const int n0 = blockIdx.y * 128;

  Frag8 acc[4][2];
#pragma unroll
  for (int mi = 0; mi < 4; ++mi)
#pragma unroll
    for (int ni = 0; ni < 2; ++ni) acc[mi][ni].v = (v8f)(0.f);

  const int steps = NHEADS * (SEQ / TK);   // 1024
  if (wid == 0) {
    tdm_load_tile(lds_off(&As[0][0]), P  + (size_t)m0 * SEQ, SEQ);
    tdm_load_tile(lds_off(&Bs[0][0]), Vt + (size_t)n0 * SEQ, SEQ);
  }
  for (int t = 0; t < steps; ++t) {
    if (wid == 0) __builtin_amdgcn_s_wait_tensorcnt(0);
    __syncthreads();
    const int cur = t & 1;
    if (wid == 0 && t + 1 < steps) {
      const int hn = (t + 1) >> 7;
      const int j0 = ((t + 1) & 127) * TK;
      tdm_load_tile(lds_off(&As[cur ^ 1][0]),
                    P + (size_t)hn * SEQ * SEQ + (size_t)m0 * SEQ + j0, SEQ);
      tdm_load_tile(lds_off(&Bs[cur ^ 1][0]),
                    Vt + (size_t)hn * NOUTD * SEQ + (size_t)n0 * SEQ + j0, SEQ);
    }
    v16bf a[4], bm[2];
#pragma unroll
    for (int mi = 0; mi < 4; ++mi) a[mi] = ldA_lds(&As[cur][0], wm * 64 + 16 * mi, lane);
#pragma unroll
    for (int ni = 0; ni < 2; ++ni) bm[ni] = ldB_lds(&Bs[cur][0], wn * 32 + 16 * ni, lane);
#pragma unroll
    for (int mi = 0; mi < 4; ++mi)
#pragma unroll
      for (int ni = 0; ni < 2; ++ni) acc[mi][ni].v = wmma_bf16(a[mi], bm[ni], acc[mi][ni].v);
  }

  const int n = lane & 15, mh = lane >> 4;
#pragma unroll
  for (int mi = 0; mi < 4; ++mi)
#pragma unroll
    for (int ni = 0; ni < 2; ++ni)
#pragma unroll
      for (int r = 0; r < 8; ++r)
        O[(size_t)(m0 + wm * 64 + 16 * mi + 8 * mh + r) * NOUTD +
          (n0 + wn * 32 + 16 * ni + n)] = acc[mi][ni].f[r];
}

// ---------------- kernel 4: row LayerNorm ----------------
__global__ __launch_bounds__(256) void layernorm(const float* __restrict__ O,
                                                 const float* __restrict__ lw,
                                                 const float* __restrict__ lb,
                                                 float* __restrict__ out) {
  const int row = blockIdx.x;
  const float* x = O + (size_t)row * NOUTD;
  float s = 0.f, s2 = 0.f;
  for (int e = threadIdx.x; e < NOUTD; e += 256) { float v = x[e]; s += v; s2 += v * v; }
#pragma unroll
  for (int off = 16; off > 0; off >>= 1) { s += __shfl_xor(s, off); s2 += __shfl_xor(s2, off); }
  __shared__ float as[8], as2[8];
  const int wid = threadIdx.x >> 5, lane = threadIdx.x & 31;
  if (lane == 0) { as[wid] = s; as2[wid] = s2; }
  __syncthreads();
  s = 0.f; s2 = 0.f;
#pragma unroll
  for (int ww = 0; ww < 8; ++ww) { s += as[ww]; s2 += as2[ww]; }
  const float mean = s * (1.0f / NOUTD);
  const float var  = s2 * (1.0f / NOUTD) - mean * mean;
  const float rstd = rsqrtf(var + 1e-5f);
  for (int e = threadIdx.x; e < NOUTD; e += 256)
    out[(size_t)row * NOUTD + e] = (x[e] - mean) * rstd * lw[e] + lb[e];
}

extern "C" void kernel_launch(void* const* d_in, const int* in_sizes, int n_in,
                              void* d_out, int out_size, void* d_ws, size_t ws_size,
                              hipStream_t stream) {
  const float* x    = (const float*)d_in[0];
  const float* W    = (const float*)d_in[1];
  const float* bias = (const float*)d_in[2];
  const float* lnw  = (const float*)d_in[3];
  const float* lnb  = (const float*)d_in[4];
  float* out = (float*)d_out;

  unsigned short* xb = (unsigned short*)d_ws;                 //  8 MB  x  bf16
  unsigned short* Wb = xb + (size_t)SEQ * NIN;                // 20 MB  W  bf16
  unsigned short* Kb = Wb + (size_t)DTOT * NIN;               //  8 MB  keys (scaled)
  unsigned short* Qb = Kb + (size_t)SEQ * NHEADS * NHD;       //  8 MB  queries
  unsigned short* Vt = Qb + (size_t)SEQ * NHEADS * NHD;       // 64 MB  V transposed [h][e][s]
  unsigned short* P  = Vt + (size_t)NHEADS * NOUTD * SEQ;     // 256 MB probs bf16 [h][i][j]
  float*          O  = (float*)(P + (size_t)NHEADS * SEQ * SEQ); // 16 MB attention output

  cvt_f32_bf16<<<(SEQ * NIN / 4 + 255) / 256, 256, 0, stream>>>(x, xb, SEQ * NIN / 4);
  cvt_f32_bf16<<<(DTOT * NIN / 4 + 255) / 256, 256, 0, stream>>>(W, Wb, DTOT * NIN / 4);
  qkv_gemm<<<dim3(SEQ / 128, DTOT / 128), 256, 0, stream>>>(xb, Wb, bias, Kb, Qb, Vt);
  attn_softmax<<<dim3(SEQ / 16, NHEADS), 256, 0, stream>>>(Kb, Qb, P);
  av_gemm<<<dim3(SEQ / 128, NOUTD / 128), 256, 0, stream>>>(P, Vt, O);
  layernorm<<<SEQ, 256, 0, stream>>>(O, lnw, lnb, out);
}
